// GCN_48438641164787
// MI455X (gfx1250) — matrix operand order repaired and checked by reading, not compiled, not run on hardware
//
#include <hip/hip_runtime.h>
#include <hip/hip_bf16.h>

// ---------------------------------------------------------------------------
// 3-layer typed GCN for MI455X (gfx1250, wave32).
//  - Dense per-type projections via V_WMMA_F32_16X16X32_BF16 (4 accumulators,
//    one per node type; label-select in epilogue).
//  - SpMM via float4 gather from L2-resident support + global_atomic_add_f32.
// Working set (~150MB) fits MI455X's 192MB L2, so propagation is L2-bound.
// ---------------------------------------------------------------------------

typedef __attribute__((ext_vector_type(16))) __bf16 v16bf;
typedef __attribute__((ext_vector_type(8)))  float  v8f;

#define NT 4  // node types

// ---- f32 -> bf16 elementwise convert ---------------------------------------
__global__ __launch_bounds__(256)
void k_f32_to_bf16(const float* __restrict__ in, __bf16* __restrict__ out, long n) {
  long i = (long)blockIdx.x * blockDim.x + threadIdx.x;
  if (i < n) out[i] = (__bf16)in[i];
}

// ---- W[t][k][o] f32  ->  Wt[t][o][k] bf16 (transpose for contiguous-K B) ---
__global__ __launch_bounds__(256)
void k_wt_bf16(const float* __restrict__ W, __bf16* __restrict__ Wt, int K, int O) {
  int i = blockIdx.x * blockDim.x + threadIdx.x;
  int total = NT * K * O;
  if (i >= total) return;
  int o = i % O;
  int k = (i / O) % K;
  int t = i / (O * K);
  Wt[((size_t)t * O + o) * (size_t)K + k] = (__bf16)W[i];
}

// ---- WMMA projection: S[n][o] = (A[n,:] @ W[label[n]])[o] ------------------
// block = (32, 4): 4 waves, each wave owns one 16x16 output tile.
__global__ __launch_bounds__(128)
void k_gemm_sel(const __bf16* __restrict__ A,     // [N, K] bf16 row-major
                const __bf16* __restrict__ Wt,    // [NT, O, K] bf16
                const int*    __restrict__ labels,
                float*        __restrict__ S,     // [N, O] f32
                int K, int O) {
  const int lane   = threadIdx.x;           // 0..31
  const int wave   = threadIdx.y;           // 0..3
  const int tile_m = blockIdx.x;
  const int tile_o = blockIdx.y * 4 + wave;
  const int half   = lane >> 4;             // 0: lanes 0-15, 1: lanes 16-31
  const int sub    = lane & 15;

  const __bf16* ap = A + (size_t)(tile_m * 16 + sub) * K;
  const size_t  ws = (size_t)O * K;         // stride between types
  const __bf16* bp = Wt + (size_t)(tile_o * 16 + sub) * K + 16 * half;

  v8f c0 = {}, c1 = {}, c2 = {}, c3 = {};

  for (int k0 = 0; k0 < K; k0 += 32) {
    union { v16bf v; unsigned u[8]; } a, b0, b1, b2, b3;
    // A-frag: lane = row, 16 contiguous K values per lane (half-selected)
    //  VGPR q -> K = {16*half+2q, 16*half+2q+1}
#pragma unroll
    for (int q = 0; q < 8; ++q) {
      int ka = k0 + 16 * half + 2 * q;
      a.u[q] = *(const unsigned*)(ap + ka);
    }
    // B-frag: lane = column, 16 contiguous K values per lane (half-selected)
#pragma unroll
    for (int q = 0; q < 8; ++q) {
      b0.u[q] = *(const unsigned*)(bp + 0 * ws + k0 + 2 * q);
      b1.u[q] = *(const unsigned*)(bp + 1 * ws + k0 + 2 * q);
      b2.u[q] = *(const unsigned*)(bp + 2 * ws + k0 + 2 * q);
      b3.u[q] = *(const unsigned*)(bp + 3 * ws + k0 + 2 * q);
    }
    c0 = __builtin_amdgcn_wmma_f32_16x16x32_bf16(false, a.v, false, b0.v, (short)0, c0, false, false);
    c1 = __builtin_amdgcn_wmma_f32_16x16x32_bf16(false, a.v, false, b1.v, (short)0, c1, false, false);
    c2 = __builtin_amdgcn_wmma_f32_16x16x32_bf16(false, a.v, false, b2.v, (short)0, c2, false, false);
    c3 = __builtin_amdgcn_wmma_f32_16x16x32_bf16(false, a.v, false, b3.v, (short)0, c3, false, false);
  }

  // C/D layout: VGPR q of lane -> row q + 8*half, col = sub
  const int col = tile_o * 16 + sub;
#pragma unroll
  for (int q = 0; q < 8; ++q) {
    int node = tile_m * 16 + q + 8 * half;
    int lab  = labels[node];
    float r  = (lab == 0) ? c0[q] : (lab == 1) ? c1[q] : (lab == 2) ? c2[q] : c3[q];
    S[(size_t)node * O + col] = r;
  }
}

// ---- agg[n][c] = b[c] (O is a power of two; omask = O-1) -------------------
__global__ __launch_bounds__(256)
void k_bias_init(float* __restrict__ agg, const float* __restrict__ b, int omask, long n) {
  long i = (long)blockIdx.x * blockDim.x + threadIdx.x;
  if (i < n) agg[i] = b[i & omask];
}

// ---- agg[dst] += w * S[src] (float4 per thread, atomic scatter) ------------
__global__ __launch_bounds__(256)
void k_spmm(const float* __restrict__ S,
            const int* __restrict__ src, const int* __restrict__ dst,
            const float* __restrict__ w,
            float* __restrict__ agg, int O, int cshift, int cmask, long total) {
  long t = (long)blockIdx.x * blockDim.x + threadIdx.x;
  if (t >= total) return;
  long e = t >> cshift;
  int  c = (int)(t & cmask) << 2;
  int  s = src[e], d = dst[e];
  float ww = w[e];
  const float4 v = *(const float4*)(S + (size_t)s * O + c);
  float* o = agg + (size_t)d * O + c;
  atomicAdd(o + 0, v.x * ww);
  atomicAdd(o + 1, v.y * ww);
  atomicAdd(o + 2, v.z * ww);
  atomicAdd(o + 3, v.w * ww);
}

// ---- h = bf16(relu(agg)) ---------------------------------------------------
__global__ __launch_bounds__(256)
void k_relu_bf16(const float* __restrict__ agg, __bf16* __restrict__ h, long n) {
  long i = (long)blockIdx.x * blockDim.x + threadIdx.x;
  if (i < n) {
    float v = agg[i];
    h[i] = (__bf16)fmaxf(v, 0.0f);
  }
}

// ---------------------------------------------------------------------------
extern "C" void kernel_launch(void* const* d_in, const int* in_sizes, int n_in,
                              void* d_out, int out_size, void* d_ws, size_t ws_size,
                              hipStream_t stream) {
  const float* x      = (const float*)d_in[0];
  const int*   esrc   = (const int*)  d_in[1];
  const int*   edst   = (const int*)  d_in[2];
  const float* ew     = (const float*)d_in[3];
  const int*   labels = (const int*)  d_in[4];
  const float* W1 = (const float*)d_in[5];
  const float* b1 = (const float*)d_in[6];
  const float* W2 = (const float*)d_in[7];
  const float* b2 = (const float*)d_in[8];
  const float* W3 = (const float*)d_in[9];
  const float* b3 = (const float*)d_in[10];

  const int  NF = 256, NH = 256, NC = 64;
  const long N  = in_sizes[4];   // labels: one per node
  const long E  = in_sizes[1];   // edge_src

  // --- workspace carving ---
  char*  ws  = (char*)d_ws;
  size_t off = 0;
  auto carve = [&](size_t bytes) -> char* {
    char* p = ws + off;
    off += (bytes + 255) & ~(size_t)255;
    return p;
  };
  __bf16* xh      = (__bf16*)carve((size_t)N * NF * sizeof(__bf16));   // activations (reused per layer)
  float*  support = (float*) carve((size_t)N * NH * sizeof(float));
  float*  agg     = (float*) carve((size_t)N * NH * sizeof(float));
  __bf16* W1t     = (__bf16*)carve((size_t)NT * NF * NH * sizeof(__bf16));
  __bf16* W2t     = (__bf16*)carve((size_t)NT * NH * NH * sizeof(__bf16));
  __bf16* W3t     = (__bf16*)carve((size_t)NT * NH * NC * sizeof(__bf16));
  (void)ws_size;

  const dim3 gblk(32, 4, 1);  // 4 waves per block, one 16x16 tile per wave

  // --- one-time-per-call conversions (deterministic, cheap) ---
  {
    long n = N * NF;
    k_f32_to_bf16<<<(unsigned)((n + 255) / 256), 256, 0, stream>>>(x, xh, n);
    int t1 = NT * NF * NH;
    k_wt_bf16<<<(t1 + 255) / 256, 256, 0, stream>>>(W1, W1t, NF, NH);
    int t2 = NT * NH * NH;
    k_wt_bf16<<<(t2 + 255) / 256, 256, 0, stream>>>(W2, W2t, NH, NH);
    int t3 = NT * NH * NC;
    k_wt_bf16<<<(t3 + 255) / 256, 256, 0, stream>>>(W3, W3t, NH, NC);
  }

  const unsigned tiles_m = (unsigned)(N / 16);  // N = 50000 -> 3125 exact

  // ---------------- Layer 1: x -> h1 ----------------
  k_gemm_sel<<<dim3(tiles_m, NH / 16 / 4, 1), gblk, 0, stream>>>(xh, W1t, labels, support, NF, NH);
  {
    long n = N * NH;
    k_bias_init<<<(unsigned)((n + 255) / 256), 256, 0, stream>>>(agg, b1, NH - 1, n);
    long total = E << 6;  // 64 float4 chunks per edge
    k_spmm<<<(unsigned)((total + 255) / 256), 256, 0, stream>>>(support, esrc, edst, ew, agg, NH, 6, 63, total);
    k_relu_bf16<<<(unsigned)((n + 255) / 256), 256, 0, stream>>>(agg, xh, n);
  }

  // ---------------- Layer 2: h1 -> h2 ----------------
  k_gemm_sel<<<dim3(tiles_m, NH / 16 / 4, 1), gblk, 0, stream>>>(xh, W2t, labels, support, NH, NH);
  {
    long n = N * NH;
    k_bias_init<<<(unsigned)((n + 255) / 256), 256, 0, stream>>>(agg, b2, NH - 1, n);
    long total = E << 6;
    k_spmm<<<(unsigned)((total + 255) / 256), 256, 0, stream>>>(support, esrc, edst, ew, agg, NH, 6, 63, total);
    k_relu_bf16<<<(unsigned)((n + 255) / 256), 256, 0, stream>>>(agg, xh, n);
  }

  // ---------------- Layer 3: h2 -> out (f32, no relu) ----------------
  float* outp = (float*)d_out;
  k_gemm_sel<<<dim3(tiles_m, 1, 1), gblk, 0, stream>>>(xh, W3t, labels, support, NH, NC);
  {
    long n = N * NC;
    k_bias_init<<<(unsigned)((n + 255) / 256), 256, 0, stream>>>(outp, b3, NC - 1, n);
    long total = E << 4;  // 16 float4 chunks per edge
    k_spmm<<<(unsigned)((total + 255) / 256), 256, 0, stream>>>(support, esrc, edst, ew, outp, NC, 4, 15, total);
  }
}
